// HistogramAndEdgeLoss_71159018160700
// MI455X (gfx1250) — compile-verified, matrix operand order checked by reading, not run on hardware
//
#include <hip/hip_runtime.h>
#include <hip/hip_bf16.h>
#include <math.h>

// ---------------------------------------------------------------------------
// HistogramAndEdgeLoss for gfx1250 (MI455X, wave32)
//   inputs : d_in[0] = fake_images [16,1,512,512] f32
//            d_in[1] = real_images [16,1,512,512] f32
//   output : d_out[0] = scalar f32 loss
// ---------------------------------------------------------------------------

#define B_    16
#define H_    512
#define W_    512
#define WS_   8
#define BINS_ 256
#define HW_   (H_ / WS_)          // 64
#define WW_   (W_ / WS_)          // 64
#define NWIN_ (HW_ * WW_)         // 4096
#define NPIX_ ((size_t)B_ * H_ * W_)  // 4194304

typedef __attribute__((ext_vector_type(16))) _Float16 v16h;
typedef __attribute__((ext_vector_type(8)))  float    v8f;

// ---- monotone float <-> uint key (deterministic integer min/max atomics) ----
__device__ __forceinline__ unsigned fkey(float f) {
  unsigned u = __float_as_uint(f);
  return (u & 0x80000000u) ? ~u : (u | 0x80000000u);
}
__device__ __forceinline__ float fdecode(unsigned k) {
  unsigned u = (k & 0x80000000u) ? (k ^ 0x80000000u) : ~k;
  return __uint_as_float(u);
}

// ---- zero-padded pixel read ----
__device__ __forceinline__ float px(const float* img, int b, int y, int x) {
  if ((unsigned)y >= (unsigned)H_ || (unsigned)x >= (unsigned)W_) return 0.0f;
  return img[((size_t)b << 18) + ((size_t)((unsigned)y) << 9) + (unsigned)x];
}

// ---- Sobel cross-correlation (matches lax.conv_general_dilated, pad=1) ----
__device__ __forceinline__ void sobel(const float* img, int b, int y, int x,
                                      float& ex, float& ey) {
  float a00 = px(img, b, y - 1, x - 1), a01 = px(img, b, y - 1, x), a02 = px(img, b, y - 1, x + 1);
  float a10 = px(img, b, y,     x - 1),                             a12 = px(img, b, y,     x + 1);
  float a20 = px(img, b, y + 1, x - 1), a21 = px(img, b, y + 1, x), a22 = px(img, b, y + 1, x + 1);
  ex = (a02 - a00) + 2.0f * (a12 - a10) + (a22 - a20);
  ey = (a20 - a00) + 2.0f * (a21 - a01) + (a22 - a02);
}

// ---- torch.histc bin index: floor((v-mn)*scale), clip, v>=mx -> last bin ----
__device__ __forceinline__ int binIdx(float v, float mn, float mx, float scale) {
  if (v >= mx) return BINS_ - 1;
  float t = (v - mn) * scale;       // t >= 0 (mn is the global minimum)
  int i = (int)t;
  if (i > BINS_ - 1) i = BINS_ - 1;
  if (i < 0) i = 0;
  return i;
}

// ---------------------------------------------------------------------------
// Kernel 0: init min/max key slots. layout: gkeys[img*6 + q*2 + {0:min,1:max}]
//           img 0 = fake, img 1 = real; q 0 = raw, 1 = mag, 2 = dir
// ---------------------------------------------------------------------------
__global__ void hel_init_kernel(unsigned* gkeys) {
  int t = threadIdx.x;
  if (t < 12) gkeys[t] = (t & 1) ? 0u : 0xFFFFFFFFu;
}

// ---------------------------------------------------------------------------
// Kernel 1: global min/max of raw / sobel-magnitude / sobel-direction per image
// grid (2048, 2) x 256 threads, 8 strided pixels each
// ---------------------------------------------------------------------------
__global__ void hel_minmax_kernel(const float* __restrict__ fake,
                                  const float* __restrict__ real,
                                  unsigned* __restrict__ gkeys) {
  const float* img = blockIdx.y ? real : fake;
  const int base   = (int)blockIdx.y * 6;
  const size_t g   = (size_t)blockIdx.x * blockDim.x + threadIdx.x;   // 524288 threads
  const size_t stride = (size_t)2048 * 256;

  float rmn =  3.0e38f, rmx = -3.0e38f;
  float mmn =  3.0e38f, mmx = -3.0e38f;
  float dmn =  3.0e38f, dmx = -3.0e38f;

  for (int j = 0; j < 8; ++j) {
    size_t p = g + (size_t)j * stride;       // covers all NPIX_ pixels
    int b = (int)(p >> 18);
    int y = (int)((p >> 9) & 511u);
    int x = (int)(p & 511u);
    float v = img[p];
    rmn = fminf(rmn, v); rmx = fmaxf(rmx, v);
    float ex, ey;
    sobel(img, b, y, x, ex, ey);
    float mg = sqrtf(ex * ex + ey * ey);
    float dr = atan2f(ey, ex);
    mmn = fminf(mmn, mg); mmx = fmaxf(mmx, mg);
    dmn = fminf(dmn, dr); dmx = fmaxf(dmx, dr);
  }

  __shared__ unsigned smin[3], smax[3];
  if (threadIdx.x < 3) { smin[threadIdx.x] = 0xFFFFFFFFu; smax[threadIdx.x] = 0u; }
  __syncthreads();
  atomicMin(&smin[0], fkey(rmn)); atomicMax(&smax[0], fkey(rmx));
  atomicMin(&smin[1], fkey(mmn)); atomicMax(&smax[1], fkey(mmx));
  atomicMin(&smin[2], fkey(dmn)); atomicMax(&smax[2], fkey(dmx));
  __syncthreads();
  if (threadIdx.x < 3) {
    atomicMin(&gkeys[base + threadIdx.x * 2 + 0], smin[threadIdx.x]);
    atomicMax(&gkeys[base + threadIdx.x * 2 + 1], smax[threadIdx.x]);
  }
}

// ---------------------------------------------------------------------------
// Kernel 2: one wave per 8x8 window. Build fake/real 256-bin LDS histograms
// (raw pass, then mag+dir pass), stage |diff| (exact integers <= 2048) as f16,
// reduce all 512 diffs with one v_wmma_f32_16x16x32_f16 (A = ones).
// ---------------------------------------------------------------------------
__global__ void __launch_bounds__(32)
hel_hist_kernel(const float* __restrict__ fake,
                const float* __restrict__ real,
                const unsigned* __restrict__ gkeys,
                float* __restrict__ partials) {
  const int win = blockIdx.x;
  const int wy  = win >> 6;
  const int wx  = win & 63;
  const int l   = threadIdx.x;              // 0..31, one wave

  __shared__ unsigned hF[BINS_], hR[BINS_];
  __shared__ __align__(32) _Float16 diffs[2 * BINS_];

  // decode per-image / per-quantity min,max,scale
  float mnF0 = fdecode(gkeys[0]),  mxF0 = fdecode(gkeys[1]);
  float mnF1 = fdecode(gkeys[2]),  mxF1 = fdecode(gkeys[3]);
  float mnF2 = fdecode(gkeys[4]),  mxF2 = fdecode(gkeys[5]);
  float mnR0 = fdecode(gkeys[6]),  mxR0 = fdecode(gkeys[7]);
  float mnR1 = fdecode(gkeys[8]),  mxR1 = fdecode(gkeys[9]);
  float mnR2 = fdecode(gkeys[10]), mxR2 = fdecode(gkeys[11]);
  float scF0 = (float)BINS_ / (mxF0 - mnF0);
  float scF1 = (float)BINS_ / (mxF1 - mnF1);
  float scF2 = (float)BINS_ / (mxF2 - mnF2);
  float scR0 = (float)BINS_ / (mxR0 - mnR0);
  float scR1 = (float)BINS_ / (mxR1 - mnR1);
  float scR2 = (float)BINS_ / (mxR2 - mnR2);

  // ---------------- phase 1: raw histogram ----------------
  for (int k = 0; k < 8; ++k) { hF[l * 8 + k] = 0u; hR[l * 8 + k] = 0u; }
  __syncthreads();

  for (int pp = 0; pp < 4; ++pp) {
    int pr = l + pp * 32;                 // (batch,row) pair id 0..127
    int b  = pr >> 3;
    int r  = pr & 7;
    int y  = wy * WS_ + r;
    const float* fb = fake + (((size_t)b << 18) + ((size_t)y << 9) + wx * WS_);
    const float* rb = real + (((size_t)b << 18) + ((size_t)y << 9) + wx * WS_);
    for (int c = 0; c < 8; ++c) {
      atomicAdd(&hF[binIdx(fb[c], mnF0, mxF0, scF0)], 1u);
      atomicAdd(&hR[binIdx(rb[c], mnR0, mxR0, scR0)], 1u);
    }
  }
  __syncthreads();
  for (int k = 0; k < 8; ++k) {
    int bin = l * 8 + k;
    float d = fabsf((float)(int)hF[bin] - (float)(int)hR[bin]);  // <= 1024, f16-exact
    diffs[bin] = (_Float16)d;
  }
  __syncthreads();

  // ---------------- phase 2: edge (mag + dir) histogram ----------------
  for (int k = 0; k < 8; ++k) { hF[l * 8 + k] = 0u; hR[l * 8 + k] = 0u; }
  __syncthreads();

  for (int pp = 0; pp < 4; ++pp) {
    int pr = l + pp * 32;
    int b  = pr >> 3;
    int r  = pr & 7;
    int y  = wy * WS_ + r;
    for (int c = 0; c < 8; ++c) {
      int x = wx * WS_ + c;
      float ex, ey;
      sobel(fake, b, y, x, ex, ey);
      float mg = sqrtf(ex * ex + ey * ey);
      float dr = atan2f(ey, ex);
      atomicAdd(&hF[binIdx(mg, mnF1, mxF1, scF1)], 1u);
      atomicAdd(&hF[binIdx(dr, mnF2, mxF2, scF2)], 1u);
      sobel(real, b, y, x, ex, ey);
      mg = sqrtf(ex * ex + ey * ey);
      dr = atan2f(ey, ex);
      atomicAdd(&hR[binIdx(mg, mnR1, mxR1, scR1)], 1u);
      atomicAdd(&hR[binIdx(dr, mnR2, mxR2, scR2)], 1u);
    }
  }
  __syncthreads();
  for (int k = 0; k < 8; ++k) {
    int bin = l * 8 + k;
    float d = fabsf((float)(int)hF[bin] - (float)(int)hR[bin]);  // <= 2048, f16-exact
    diffs[BINS_ + bin] = (_Float16)d;
  }
  __syncthreads();

  // ---------------- WMMA reduction of the 512 f16 diffs ----------------
  // D = ones(16x32) * B(32x16) + 0  =>  row 0 of D = column sums of B.
  // Sum of row 0 across N = grand total of all 512 staged values,
  // independent of the exact (k,n) lane mapping of B.
  v16h av;
  for (int i = 0; i < 16; ++i) av[i] = (_Float16)1.0f;
  v16h bv = *(const v16h*)(&diffs[l * 16]);   // 32 lanes x 16 halves = 512 vals
  v8f  cv = {};
  cv = __builtin_amdgcn_wmma_f32_16x16x32_f16(
      /*neg_a=*/false, av, /*neg_b=*/false, bv,
      /*c_mod=*/(short)0, cv, /*reuse_a=*/false, /*reuse_b=*/false);

  float t = cv[0];                 // lanes 0..15 hold D[0][n], n = lane
  t += __shfl_xor(t, 8, 16);
  t += __shfl_xor(t, 4, 16);
  t += __shfl_xor(t, 2, 16);
  t += __shfl_xor(t, 1, 16);
  if (l == 0) partials[win] = t;   // per-window raw-diff + edge-diff sum
}

// ---------------------------------------------------------------------------
// Kernel 3: deterministic fixed-order reduction of 4096 partials -> scalar
// loss = (sum raw diffs + sum edge diffs) / (64*64*256)
// ---------------------------------------------------------------------------
__global__ void hel_final_kernel(const float* __restrict__ partials,
                                 float* __restrict__ out) {
  __shared__ float red[256];
  int t = threadIdx.x;
  float s = 0.0f;
  for (int i = 0; i < 16; ++i) s += partials[t * 16 + i];
  red[t] = s;
  __syncthreads();
  for (int off = 128; off > 0; off >>= 1) {
    if (t < off) red[t] += red[t + off];
    __syncthreads();
  }
  if (t == 0) out[0] = red[0] * (1.0f / (float)(NWIN_ * BINS_));
}

// ---------------------------------------------------------------------------
extern "C" void kernel_launch(void* const* d_in, const int* in_sizes, int n_in,
                              void* d_out, int out_size, void* d_ws, size_t ws_size,
                              hipStream_t stream) {
  (void)in_sizes; (void)n_in; (void)out_size; (void)ws_size;
  const float* fake = (const float*)d_in[0];
  const float* real = (const float*)d_in[1];
  float* out = (float*)d_out;

  unsigned* gkeys   = (unsigned*)d_ws;                    // 12 uints
  float*    partials = (float*)((char*)d_ws + 256);       // 4096 floats

  hel_init_kernel<<<1, 32, 0, stream>>>(gkeys);
  hel_minmax_kernel<<<dim3(2048, 2), 256, 0, stream>>>(fake, real, gkeys);
  hel_hist_kernel<<<NWIN_, 32, 0, stream>>>(fake, real, gkeys, partials);
  hel_final_kernel<<<1, 256, 0, stream>>>(partials, out);
}
